// PACEConv2d_Split2x_39651138077502
// MI455X (gfx1250) — matrix operand order repaired and checked by reading, not compiled, not run on hardware
//
#include <hip/hip_runtime.h>

// ---------------------------------------------------------------------------
// Factorized spectral conv (FNO-style) for gfx1250 / MI455X.
//   pass over fast axis:  Y = irfft32( per-mode-mix( rfft32(X) ) )
// expressed as 3 GEMMs on 16x16 WMMA tiles (f16 in, f32 accumulate).
// One kernel, two launches: pass1 (W axis, w2) -> ws transposed,
//                           pass2 (H axis, w1) -> out transposed back.
// ---------------------------------------------------------------------------

typedef _Float16 half_t;
typedef __attribute__((ext_vector_type(16))) _Float16 v16h;
typedef __attribute__((ext_vector_type(8)))  _Float16 v8h;
typedef __attribute__((ext_vector_type(8)))  float    v8f;
typedef __attribute__((ext_vector_type(4)))  float    v4f;

#define NFFT   256
#define NMODE  32
#define HB     16          // spatial rows per workgroup
#define CI     32
#define CO     32
#define WAVES  8
#define TPB    (WAVES * 32)

// ---- LDS layout (bytes). Phases overlap dead regions. Total 256 KB. ----
#define OFF_FMIX   0            // [32 k][16 h][64 c(re|im)] f16   = 64 KB (A->B)
#define OFF_GMIX   (64 * 1024)  // [512 (o,h)][64 kc] f16          = 64 KB (B->C)
#define OFF_EB     (64 * 1024)  // phase A scratch: [256 w][64 kc] = 32 KB
#define OFF_XSTAGE (96 * 1024)  // phase A scratch: 8 x [16][256]  = 64 KB
#define OFF_WRE    (128 * 1024) // phase B: [32 k][32 i][32 o] f16 = 64 KB
#define OFF_WIM    (192 * 1024) //                                 = 64 KB
#define OFF_CINV   0            // phase C: [64 kc][256 w] f16     = 32 KB
#define SMEM_BYTES (256 * 1024)

__device__ __forceinline__ v8f wmma_f16(v16h a, v16h b, v8f c) {
    return __builtin_amdgcn_wmma_f32_16x16x32_f16(false, a, false, b,
                                                  (short)0, c, false, false);
}

// A fragment (16x32, f16): lane&15 = M row; halves[0..7] = K k0+8*(lane>>4)+0..7,
// halves[8..15] = K k0+16+8*(lane>>4)+0..7.  `base` is row-major [rows][ld].
__device__ __forceinline__ v16h load_afrag(const half_t* base, int row0, int ld, int k0) {
    const int lane = threadIdx.x & 31;
    const half_t* p = base + (row0 + (lane & 15)) * ld + k0 + ((lane >> 4) << 3);
    v8h lo = *(const v8h*)p;
    v8h hi = *(const v8h*)(p + 16);
    v16h f;
#pragma unroll
    for (int i = 0; i < 8; ++i) { f[i] = lo[i]; f[i + 8] = hi[i]; }
    return f;
}

// B fragment (32x16, f16): lane&15 = N col n0+(lane&15); halves[0..7] = K k0+8*(lane>>4)+0..7,
// halves[8..15] = K k0+16+8*(lane>>4)+0..7.
__device__ __forceinline__ v16h load_bfrag(const half_t* base, int k0, int ldb, int n0) {
    const int lane = threadIdx.x & 31;
    const half_t* p = base + (k0 + ((lane >> 4) << 3)) * ldb + n0 + (lane & 15);
    v16h f;
#pragma unroll
    for (int i = 0; i < 8; ++i) { f[i] = p[i * ldb]; f[i + 8] = p[(i + 16) * ldb]; }
    return f;
}

__global__ __launch_bounds__(TPB, 1)
void spectral_pass(const float* __restrict__ in,
                   const float* __restrict__ wre,
                   const float* __restrict__ wim,
                   float* __restrict__ out)
{
    extern __shared__ char smem[];
    half_t* sF   = (half_t*)(smem + OFF_FMIX);
    half_t* sG   = (half_t*)(smem + OFF_GMIX);
    half_t* sE   = (half_t*)(smem + OFF_EB);
    half_t* sX   = (half_t*)(smem + OFF_XSTAGE);
    half_t* sWre = (half_t*)(smem + OFF_WRE);
    half_t* sWim = (half_t*)(smem + OFF_WIM);
    half_t* sC   = (half_t*)(smem + OFF_CINV);

    const int tid  = threadIdx.x;
    const int lane = tid & 31;
    const int wave = tid >> 5;
    const int hb = blockIdx.x, g = blockIdx.y, b = blockIdx.z;
    const int h0 = hb * HB;
    const long chanBase = (long)(b * 64 + g * 32);

    const float STEP = 6.2831853071795864f / 256.0f;  // 2*pi/N
    const float FWD  = 1.0f / 16.0f;                  // 1/sqrt(256), ortho

    // ---- forward DFT table  sE[w][kc] : kc<32 -> cos, kc>=32 -> -sin ----
    for (int d = tid; d < 256 * 64; d += TPB) {
        int w = d >> 6, kc = d & 63, k = kc & 31;
        float th = (float)((k * w) & 255) * STEP;     // exact mod-256 reduction
        float v = (kc < 32) ? __cosf(th) : -__sinf(th);
        sE[d] = (half_t)(v * FWD);
    }
    __syncthreads();

    // ---- Phase A: per channel i, F(16h x 64kc) = X(16h x 256w) @ E ----
    half_t* myX = sX + wave * (HB * 256);
    for (int ii = 0; ii < CI / WAVES; ++ii) {
        const int i = wave + WAVES * ii;
        const float* src = in + ((chanBase + i) * 256 + h0) * 256;
        for (int t4 = lane * 4; t4 < HB * 256; t4 += 32 * 4) {   // stage f32->f16
            int r = t4 >> 8, w = t4 & 255;
            v4f x4 = *(const v4f*)(src + r * 256 + w);
            half_t* dp = myX + r * 256 + w;
            dp[0] = (half_t)x4[0]; dp[1] = (half_t)x4[1];
            dp[2] = (half_t)x4[2]; dp[3] = (half_t)x4[3];
        }
        // prefetch next channel's tile (16 KB) while this tile's WMMAs run
        if (ii + 1 < CI / WAVES) {
            const float* nxt = in + ((chanBase + i + WAVES) * 256 + h0) * 256;
            __builtin_prefetch(nxt + lane * 128, 0, 3);          // 512B stride
            __builtin_prefetch(nxt + lane * 128 + 64, 0, 3);
        }
        v8f acc[4] = {};
        for (int s = 0; s < 8; ++s) {                 // K = 256 in 8 steps
            v16h a = load_afrag(myX, 0, 256, s * 32);
#pragma unroll
            for (int nt = 0; nt < 4; ++nt) {
                v16h bf = load_bfrag(sE, s * 32, 64, nt * 16);
                acc[nt] = wmma_f16(a, bf, acc[nt]);
            }
        }
        const int col = lane & 15, hh8 = (lane >> 4) * 8;
#pragma unroll
        for (int nt = 0; nt < 4; ++nt) {
            int kc = nt * 16 + col;
#pragma unroll
            for (int j = 0; j < 8; ++j) {
                int h = hh8 + j;
                float v = acc[nt][j];
                if (kc < 32) sF[(kc * HB + h) * 64 + i]             = (half_t)v;
                else         sF[((kc - 32) * HB + h) * 64 + 32 + i] = (half_t)v;
            }
        }
    }
    __syncthreads();

    // ---- weights for this group: sWre/sWim [k][i][o] (B layout) ----
    const float* wreg = wre + (size_t)g * 32 * 32 * 32;
    const float* wimg = wim + (size_t)g * 32 * 32 * 32;
    for (int d = tid; d < 32 * 32 * 32; d += TPB) {
        int k = d >> 10, i = (d >> 5) & 31, o = d & 31;
        size_t s = ((size_t)i * 32 + o) * 32 + k;     // w[g][i][o][k]
        sWre[d] = (half_t)wreg[s];
        sWim[d] = (half_t)wimg[s];
    }
    __syncthreads();

    // ---- Phase B: per-mode complex mix  G[h,o] = F[h,i] @ W[i,o,k] ----
    for (int kk = 0; kk < NMODE / WAVES; ++kk) {
        const int k = wave + WAVES * kk;
        const half_t* fk = sF + k * HB * 64;
        v16h aRe = load_afrag(fk, 0, 64, 0);
        v16h aIm = load_afrag(fk, 0, 64, 32);
        v16h aNi;
#pragma unroll
        for (int t = 0; t < 16; ++t) aNi[t] = -aIm[t];
        const int col = lane & 15, hh8 = (lane >> 4) * 8;
#pragma unroll
        for (int nt = 0; nt < 2; ++nt) {
            v16h bRe = load_bfrag(sWre + k * 1024, 0, 32, nt * 16);
            v16h bIm = load_bfrag(sWim + k * 1024, 0, 32, nt * 16);
            v8f gr = {}, gi = {};
            gr = wmma_f16(aRe, bRe, gr);              // Fre*Wre - Fim*Wim
            gr = wmma_f16(aNi, bIm, gr);
            gi = wmma_f16(aRe, bIm, gi);              // Fre*Wim + Fim*Wre
            gi = wmma_f16(aIm, bRe, gi);
            int o = nt * 16 + col;
#pragma unroll
            for (int j = 0; j < 8; ++j) {
                int h = hh8 + j;
                sG[(o * HB + h) * 64 + k]      = (half_t)gr[j];
                sG[(o * HB + h) * 64 + 32 + k] = (half_t)gi[j];
            }
        }
    }
    __syncthreads();

    // ---- inverse table sC[kc][w]: a_k*cos / -a_k*sin, bin-0 imag dropped ----
    for (int d = tid; d < 64 * 256; d += TPB) {
        int kc = d >> 8, w = d & 255, k = kc & 31;
        float th = (float)((k * w) & 255) * STEP;
        float a = (k == 0) ? FWD : 2.0f * FWD;
        float v = (kc < 32) ? a * __cosf(th) : ((k == 0) ? 0.0f : -a * __sinf(th));
        sC[d] = (half_t)v;
    }
    __syncthreads();

    // ---- Phase C: Y(16h x 256w) = [Gre|Gim](16 x 64) @ C, transposed store.
    // Loop-interchanged: hoist all A fragments, load each B fragment once. ----
    {
        v16h a0[CO / WAVES], a1[CO / WAVES];
#pragma unroll
        for (int oo = 0; oo < CO / WAVES; ++oo) {
            const half_t* grow = sG + ((wave + WAVES * oo) * HB) * 64;
            a0[oo] = load_afrag(grow, 0, 64, 0);
            a1[oo] = load_afrag(grow, 0, 64, 32);
        }
        const int col = lane & 15, hh8 = (lane >> 4) * 8;
        for (int nt = 0; nt < 16; ++nt) {
            v16h b0 = load_bfrag(sC, 0, 256, nt * 16);
            v16h b1 = load_bfrag(sC, 32, 256, nt * 16);
            int w = nt * 16 + col;
#pragma unroll
            for (int oo = 0; oo < CO / WAVES; ++oo) {
                const int o = wave + WAVES * oo;
                v8f acc = {};
                acc = wmma_f16(a0[oo], b0, acc);
                acc = wmma_f16(a1[oo], b1, acc);
                float* p = out + ((long)((b * 2 + g) * 32 + o)) * 65536
                               + (long)w * 256 + h0 + hh8;   // transposed: h fastest
                v4f lo4 = {acc[0], acc[1], acc[2], acc[3]};
                v4f hi4 = {acc[4], acc[5], acc[6], acc[7]};
                *(v4f*)p = lo4;
                *(v4f*)(p + 4) = hi4;
            }
        }
    }
}

extern "C" void kernel_launch(void* const* d_in, const int* in_sizes, int n_in,
                              void* d_out, int out_size, void* d_ws, size_t ws_size,
                              hipStream_t stream) {
    const float* x    = (const float*)d_in[0];
    const float* w1re = (const float*)d_in[1];
    const float* w1im = (const float*)d_in[2];
    const float* w2re = (const float*)d_in[3];
    const float* w2im = (const float*)d_in[4];
    float* out = (float*)d_out;
    float* ws  = (float*)d_ws;   // needs 8*64*256*256 floats = 128 MiB

    dim3 grid(256 / HB, 2, 8);   // (h-block, group, batch) : 256 workgroups
    dim3 block(TPB);

    // pass 1: spectral conv along W with w2; writes [b,g,o][w][h] into ws
    spectral_pass<<<grid, block, SMEM_BYTES, stream>>>(x, w2re, w2im, ws);
    // pass 2: same operator along H (now the fast axis) with w1; transposed
    // store restores [b,g,o][h][w] into d_out
    spectral_pass<<<grid, block, SMEM_BYTES, stream>>>(ws, w1re, w1im, out);
}
